// PDELevel_32590211842054
// MI455X (gfx1250) — matrix operand order repaired and checked by reading, not compiled, not run on hardware
//
#include <hip/hip_runtime.h>
#include <math.h>

typedef __attribute__((ext_vector_type(16))) __bf16 v16bf;
typedef __attribute__((ext_vector_type(8)))  __bf16 v8bf;
typedef __attribute__((ext_vector_type(8)))  float  v8f;

#define BB   8
#define CC   256
#define DH   512
#define HW   4096
#define NPIX (BB * HW)   // 32768

#define PAD1 272   // 256 + 16 bf16 -> 544B row stride (32B multiple)
#define PAD2 528   // 512 + 16 bf16 -> 1056B row stride (32B multiple)

// ---------- helpers ----------
__device__ __forceinline__ unsigned short f2bf(float f) {
    unsigned int u = __float_as_uint(f);
    unsigned int r = u + 0x7FFFu + ((u >> 16) & 1u);   // RNE
    return (unsigned short)(r >> 16);
}
__device__ __forceinline__ int refl64(int i) {        // reflect pad, n=64
    i = (i < 0) ? -i : i;
    return (i > 63) ? (126 - i) : i;
}
__device__ __forceinline__ float sigm(float x) { return 1.0f / (1.0f + expf(-x)); }

// async global->LDS copy of 16 bytes (ASYNCcnt-tracked), CDNA5 VGLOBAL op
__device__ __forceinline__ void async_g2l_b128(const void* gptr, unsigned lds_byte_off) {
    asm volatile("global_load_async_to_lds_b128 %0, %1, off"
                 :: "v"(lds_byte_off), "v"(gptr) : "memory");
}
__device__ __forceinline__ void wait_async0() {
    asm volatile("s_wait_asynccnt 0x0" ::: "memory");
}
__device__ __forceinline__ unsigned lds_off(const void* p) {
    return (unsigned)(unsigned long long)p;  // low 32 bits of generic LDS ptr = LDS byte offset
}

// Build a 16x32 bf16 A operand in contiguous k order: lo-lanes K[0..15], hi-lanes K[16..31]
__device__ __forceinline__ v16bf load_a(const unsigned short* row, int k0, int hi) {
    const v8bf lo = *(const v8bf*)(row + k0 + hi * 16);
    const v8bf hp = *(const v8bf*)(row + k0 + hi * 16 + 8);
    return __builtin_shufflevector(lo, hp, 0,1,2,3,4,5,6,7,8,9,10,11,12,13,14,15);
}

// ---------- one-time prep: transpose weights to bf16, precompute diffusion coeffs ----------
__global__ void prep_kernel(const float* __restrict__ w1, const float* __restrict__ w2,
                            const float* __restrict__ dl,
                            unsigned short* __restrict__ w1T, unsigned short* __restrict__ w2T,
                            float* __restrict__ coeff) {
    int tid = blockIdx.x * 256 + threadIdx.x;          // 0..131071
    { int e = tid >> 8, c = tid & 255; w1T[tid] = f2bf(w1[c * DH + e]); }   // w1T[e][c]
    { int c = tid >> 9, e = tid & 511; w2T[tid] = f2bf(w2[e * CC + c]); }   // w2T[c][e]
    if (tid < 3 * CC) coeff[tid] = 0.25f * sigm(dl[tid]);
}

// ---------- pooled = mean over H*W ----------
__global__ void pool_kernel(const float* __restrict__ u, float* __restrict__ pooled) {
    __shared__ float red[256];
    int bc = blockIdx.x, t = threadIdx.x;
    const float* p = u + (size_t)bc * HW;
    float s = 0.f;
    for (int i = t; i < HW; i += 256) s += p[i];
    red[t] = s; __syncthreads();
    for (int w = 128; w > 0; w >>= 1) { if (t < w) red[t] += red[t + w]; __syncthreads(); }
    if (t == 0) pooled[bc] = red[0] * (1.0f / HW);
}

// ---------- gated memory h = h*f + i*c  (tiny 256x256 matmuls) ----------
__global__ void gates_kernel(const float* __restrict__ pooled,
                             const float* __restrict__ wf, const float* __restrict__ bfv,
                             const float* __restrict__ wi, const float* __restrict__ biv,
                             const float* __restrict__ wc, const float* __restrict__ bcv,
                             float* __restrict__ h, int t) {
    __shared__ float ps[256];
    int b = blockIdx.x, c = threadIdx.x;
    ps[c] = pooled[b * CC + c]; __syncthreads();
    float zf = bfv[c], zi = biv[c], zc = bcv[c];
    for (int k = 0; k < CC; ++k) {
        float pv = ps[k];
        zf += pv * wf[k * CC + c];
        zi += pv * wi[k * CC + c];
        zc += pv * wc[k * CC + c];
    }
    float f = sigm(zf), i_ = sigm(zi), cg = tanhf(zc);
    float hold = (t == 0) ? 0.f : h[b * CC + c];
    h[b * CC + c] = hold * f + i_ * cg;
}

// ---------- transpose + convert:  u (B,C,HW) f32 -> uT (B*HW, C) bf16 ----------
__global__ void tconv_kernel(const float* __restrict__ u, unsigned short* __restrict__ uT) {
    __shared__ float tile[32][33];
    int b = blockIdx.z, c0 = blockIdx.y * 32, p0 = blockIdx.x * 32;
    int tx = threadIdx.x, ty = threadIdx.y;
    tile[ty][tx] = u[((size_t)(b * CC + c0 + ty)) * HW + p0 + tx];
    __syncthreads();
    uT[((size_t)(b * HW + p0 + ty)) * CC + c0 + tx] = f2bf(tile[tx][ty]);
}

// ---------- GEMM1: hdnT = gelu(w1T @ uT^T)^T  (M=512,K=256,N=32768), bf16 WMMA ----------
// 512 threads = 16 waves; wave w owns rows [32w, 32w+32); B tile (64 pixels x 256 K) staged in LDS.
__global__ __launch_bounds__(512)
__attribute__((amdgpu_waves_per_eu(1)))
void gemm1_kernel(const unsigned short* __restrict__ w1T,
                  const unsigned short* __restrict__ uT,
                  const float* __restrict__ b1,
                  unsigned short* __restrict__ hdnT) {
    __shared__ __align__(32) unsigned short btile[64 * PAD1];

    const int wave = threadIdx.x >> 5, lane = threadIdx.x & 31;
    const int l15 = lane & 15, hi = lane >> 4;
    const int m_base = wave * 32;
    const int n_base = blockIdx.x * 64;                // global pixel rows (b*HW+p)

    // stage B tile: 64 rows x 256 bf16, async global->LDS, 16B chunks
    const unsigned short* gsrc = uT + (size_t)n_base * CC;
    #pragma unroll
    for (int q0 = 0; q0 < 64 * 32; q0 += 512) {        // 2048 chunks of 8 bf16
        int q = q0 + threadIdx.x;
        int row = q >> 5, c8 = (q & 31) * 8;
        async_g2l_b128(gsrc + (size_t)row * CC + c8,
                       lds_off(&btile[row * PAD1 + c8]));
    }
    __builtin_prefetch(w1T + (size_t)(m_base + l15) * CC, 0, 0);
    wait_async0();
    __syncthreads();

    v8f acc[2][4];
    #pragma unroll
    for (int mt = 0; mt < 2; ++mt)
        #pragma unroll
        for (int nt = 0; nt < 4; ++nt)
            #pragma unroll
            for (int r = 0; r < 8; ++r) acc[mt][nt][r] = 0.f;

    for (int k0 = 0; k0 < CC; k0 += 32) {
        v16bf a[2], bm[4];
        #pragma unroll
        for (int mt = 0; mt < 2; ++mt)
            a[mt] = load_a(w1T + (size_t)(m_base + mt * 16 + l15) * CC, k0, hi);
        #pragma unroll
        for (int nt = 0; nt < 4; ++nt)
            bm[nt] = *(const v16bf*)(&btile[(nt * 16 + l15) * PAD1 + k0 + hi * 16]);
        #pragma unroll
        for (int mt = 0; mt < 2; ++mt)
            #pragma unroll
            for (int nt = 0; nt < 4; ++nt)
                acc[mt][nt] = __builtin_amdgcn_wmma_f32_16x16x32_bf16(
                    false, a[mt], false, bm[nt], (short)0, acc[mt][nt], false, false);
    }

    #pragma unroll
    for (int mt = 0; mt < 2; ++mt) {
        const int e0 = m_base + mt * 16 + hi * 8;
        float bias[8];
        #pragma unroll
        for (int r = 0; r < 8; ++r) bias[r] = b1[e0 + r];
        #pragma unroll
        for (int nt = 0; nt < 4; ++nt) {
            const int pix = n_base + nt * 16 + l15;
            union { unsigned short us[8]; uint4 q; } pk;
            #pragma unroll
            for (int r = 0; r < 8; ++r) {
                float x = acc[mt][nt][r] + bias[r];
                float g = 0.5f * x * (1.0f + erff(x * 0.70710678118654752f)); // exact GELU
                pk.us[r] = f2bf(g);
            }
            *(uint4*)(hdnT + (size_t)pix * DH + e0) = pk.q;   // transposed bf16 store
        }
    }
}

// ---------- GEMM2 fused: react + laplacian + forcing + memory + Euler + optional RMSNorm ----------
__global__ __launch_bounds__(512)
__attribute__((amdgpu_waves_per_eu(1)))
void gemm2_kernel(const unsigned short* __restrict__ w2T,
                  const unsigned short* __restrict__ hdnT,
                  const float* __restrict__ b2,
                  const float* __restrict__ u_cur,
                  const float* __restrict__ forcing,
                  const float* __restrict__ h,
                  const float* __restrict__ coeff,
                  const float* __restrict__ nscale,
                  const float* __restrict__ p_logdt,
                  const float* __restrict__ p_amem,
                  const float* __restrict__ p_aforce,
                  float* __restrict__ u_out, int do_norm) {
    __shared__ __align__(32) unsigned short btile[64 * PAD2];
    __shared__ float s2[64];

    const int wave = threadIdx.x >> 5, lane = threadIdx.x & 31;
    const int l15 = lane & 15, hi = lane >> 4;
    const int b = blockIdx.y;
    const int n_base = blockIdx.x * 64;    // pixel within batch
    const int m0 = wave * 16;              // 16 waves cover 256 channels

    // stage B tile: 64 rows x 512 bf16 of hdnT, async global->LDS
    const unsigned short* hb = hdnT + (size_t)(b * HW + n_base) * DH;
    #pragma unroll
    for (int q0 = 0; q0 < 64 * 64; q0 += 512) {        // 4096 chunks of 8 bf16
        int q = q0 + threadIdx.x;
        int row = q >> 6, c8 = (q & 63) * 8;
        async_g2l_b128(hb + (size_t)row * DH + c8,
                       lds_off(&btile[row * PAD2 + c8]));
    }
    __builtin_prefetch(w2T + (size_t)(m0 + l15) * DH, 0, 0);
    wait_async0();
    __syncthreads();

    v8f acc[4];
    #pragma unroll
    for (int nt = 0; nt < 4; ++nt)
        #pragma unroll
        for (int r = 0; r < 8; ++r) acc[nt][r] = 0.f;

    const unsigned short* arow = w2T + (size_t)(m0 + l15) * DH;
    for (int k0 = 0; k0 < DH; k0 += 32) {
        v16bf a = load_a(arow, k0, hi);
        #pragma unroll
        for (int nt = 0; nt < 4; ++nt) {
            v16bf bm = *(const v16bf*)(&btile[(nt * 16 + l15) * PAD2 + k0 + hi * 16]);
            acc[nt] = __builtin_amdgcn_wmma_f32_16x16x32_bf16(
                false, a, false, bm, (short)0, acc[nt], false, false);
        }
    }

    float dt = expf(*p_logdt); dt = fminf(fmaxf(dt, 0.01f), 0.3f);
    const float am = *p_amem, af = *p_aforce;
    const int c0 = m0 + hi * 8;

    float b2v[8], hv[8], k1[8], k2[8], k4[8];
    #pragma unroll
    for (int r = 0; r < 8; ++r) {
        int c = c0 + r;
        b2v[r] = b2[c]; hv[r] = h[b * CC + c];
        k1[r] = coeff[c]; k2[r] = coeff[CC + c]; k4[r] = coeff[2 * CC + c];
    }

    float unew[4][8];
    #pragma unroll
    for (int nt = 0; nt < 4; ++nt) {
        const int p = n_base + nt * 16 + l15;
        const int x = p & 63, y = p >> 6;
        const int xm1 = refl64(x - 1), xp1 = refl64(x + 1);
        const int ym1 = refl64(y - 1), yp1 = refl64(y + 1);
        const int xm2 = refl64(x - 2), xp2 = refl64(x + 2);
        const int ym2 = refl64(y - 2), yp2 = refl64(y + 2);
        const int xm4 = refl64(x - 4), xp4 = refl64(x + 4);
        const int ym4 = refl64(y - 4), yp4 = refl64(y + 4);
        #pragma unroll
        for (int r = 0; r < 8; ++r) {
            const int c = c0 + r;
            const size_t base = (size_t)(b * CC + c) * HW;
            const float* uc = u_cur + base;
            const float ctr = uc[p];
            const int yo = y * 64;
            float lap1 = uc[yo + xm1] + uc[yo + xp1] + uc[ym1 * 64 + x] + uc[yp1 * 64 + x] - 4.f * ctr;
            float lap2 = uc[yo + xm2] + uc[yo + xp2] + uc[ym2 * 64 + x] + uc[yp2 * 64 + x] - 4.f * ctr;
            float lap4 = uc[yo + xm4] + uc[yo + xp4] + uc[ym4 * 64 + x] + uc[yp4 * 64 + x] - 4.f * ctr;
            float diff = k1[r] * lap1 + k2[r] * lap2 + k4[r] * lap4;
            float react = acc[nt][r] + b2v[r];
            float du = diff + react + am * hv[r] + af * forcing[base + p];
            unew[nt][r] = ctr + dt * du;
        }
    }

    if (do_norm) {
        if (threadIdx.x < 64) s2[threadIdx.x] = 0.f;
        __syncthreads();
        #pragma unroll
        for (int nt = 0; nt < 4; ++nt) {
            float s = 0.f;
            #pragma unroll
            for (int r = 0; r < 8; ++r) s += unew[nt][r] * unew[nt][r];
            atomicAdd(&s2[nt * 16 + l15], s);   // ds_add_f32
        }
        __syncthreads();
        #pragma unroll
        for (int nt = 0; nt < 4; ++nt) {
            float rms = sqrtf(s2[nt * 16 + l15]) * 0.0625f;  // 1/sqrt(256)
            float inv = 1.0f / (rms + 1e-6f);
            #pragma unroll
            for (int r = 0; r < 8; ++r)
                unew[nt][r] = unew[nt][r] * inv * nscale[c0 + r];
        }
    }

    #pragma unroll
    for (int nt = 0; nt < 4; ++nt) {
        const int p = n_base + nt * 16 + l15;
        #pragma unroll
        for (int r = 0; r < 8; ++r)
            u_out[(size_t)(b * CC + c0 + r) * HW + p] = unew[nt][r];
    }
}

// ---------- workspace layout (bytes) ----------
#define OFF_U0     ((size_t)0)
#define OFF_U1     ((size_t)33554432)
#define OFF_UT     ((size_t)67108864)
#define OFF_HDNT   ((size_t)83886080)
#define OFF_W1T    ((size_t)117440512)
#define OFF_W2T    ((size_t)117702656)
#define OFF_COEFF  ((size_t)117964800)
#define OFF_POOL   ((size_t)117968896)
#define OFF_H      ((size_t)117977088)

extern "C" void kernel_launch(void* const* d_in, const int* in_sizes, int n_in,
                              void* d_out, int out_size, void* d_ws, size_t ws_size,
                              hipStream_t stream) {
    const float* u       = (const float*)d_in[0];
    const float* forcing = (const float*)d_in[1];
    const float* dl      = (const float*)d_in[2];
    const float* w1      = (const float*)d_in[3];
    const float* b1      = (const float*)d_in[4];
    const float* w2      = (const float*)d_in[5];
    const float* b2      = (const float*)d_in[6];
    const float* wf      = (const float*)d_in[7];
    const float* bfv     = (const float*)d_in[8];
    const float* wi      = (const float*)d_in[9];
    const float* biv     = (const float*)d_in[10];
    const float* wc      = (const float*)d_in[11];
    const float* bcv     = (const float*)d_in[12];
    const float* nsc     = (const float*)d_in[13];
    const float* logdt   = (const float*)d_in[14];
    const float* amem    = (const float*)d_in[15];
    const float* aforce  = (const float*)d_in[16];

    char* ws = (char*)d_ws;
    float*          u0    = (float*)(ws + OFF_U0);
    float*          u1    = (float*)(ws + OFF_U1);
    unsigned short* uT    = (unsigned short*)(ws + OFF_UT);
    unsigned short* hdnT  = (unsigned short*)(ws + OFF_HDNT);
    unsigned short* w1T   = (unsigned short*)(ws + OFF_W1T);
    unsigned short* w2T   = (unsigned short*)(ws + OFF_W2T);
    float*          coeff = (float*)(ws + OFF_COEFF);
    float*          pooled= (float*)(ws + OFF_POOL);
    float*          hbuf  = (float*)(ws + OFF_H);

    hipMemcpyAsync(u0, u, (size_t)BB * CC * HW * sizeof(float),
                   hipMemcpyDeviceToDevice, stream);
    prep_kernel<<<512, 256, 0, stream>>>(w1, w2, dl, w1T, w2T, coeff);

    for (int t = 0; t < 6; ++t) {
        float* ucur  = (t & 1) ? u1 : u0;
        float* unext = (t == 5) ? (float*)d_out : ((t & 1) ? u0 : u1);
        int do_norm  = ((t + 1) % 2 == 0) ? 1 : 0;

        pool_kernel <<<BB * CC, 256, 0, stream>>>(ucur, pooled);
        gates_kernel<<<BB, 256, 0, stream>>>(pooled, wf, bfv, wi, biv, wc, bcv, hbuf, t);
        tconv_kernel<<<dim3(HW / 32, CC / 32, BB), dim3(32, 32), 0, stream>>>(ucur, uT);
        gemm1_kernel<<<dim3(NPIX / 64, 1), 512, 0, stream>>>(w1T, uT, b1, hdnT);
        gemm2_kernel<<<dim3(HW / 64, BB), 512, 0, stream>>>(
            w2T, hdnT, b2, ucur, forcing, hbuf, coeff, nsc,
            logdt, amem, aforce, unext, do_norm);
    }
}
